// GPT2LowRankAttention_55181739819318
// MI455X (gfx1250) — matrix operand order repaired and checked by reading, not compiled, not run on hardware
//
#include <hip/hip_runtime.h>
#include <stdint.h>

typedef __bf16 bf16_t;
typedef __attribute__((ext_vector_type(16))) __bf16 v16bf;
typedef __attribute__((ext_vector_type(8)))  __bf16 v8bf;
typedef __attribute__((ext_vector_type(8)))  float   v8f;

namespace {

constexpr int Tn  = 4096;
constexpr int Dn  = 1024;
constexpr int Hn  = 16;
constexpr int Rn  = 64;
constexpr int HDn = 64;

union V16U { v16bf v; v8bf h[2]; };

__device__ __forceinline__ v8f wmma_bf16(v16bf a, v16bf b, v8f c) {
  // D = A(16x32 bf16) * B(32x16 bf16) + C(16x16 f32)
  return __builtin_amdgcn_wmma_f32_16x16x32_bf16(false, a, false, b, (short)0, c,
                                                 false, false);
}

// CDNA5 async DMA: copy 16B/lane from global straight into LDS (ASYNCcnt).
__device__ __forceinline__ void async_load_b128(unsigned lds_off, const void* gaddr) {
  asm volatile("global_load_async_to_lds_b128 %0, %1, off"
               :: "v"(lds_off), "v"(gaddr) : "memory");
}
__device__ __forceinline__ void wait_async0() {
  asm volatile("s_wait_asynccnt 0x0" ::: "memory");
}
__device__ __forceinline__ unsigned lds_off_of(const void* p) {
  // flat LDS addresses carry the workgroup-relative offset in addr[31:0]
  return (unsigned)(uintptr_t)p;
}

__device__ __forceinline__ float rmax16(float v) {
#pragma unroll
  for (int m = 1; m < 16; m <<= 1) v = fmaxf(v, __shfl_xor(v, m, 32));
  return v;
}
__device__ __forceinline__ float rsum16(float v) {
#pragma unroll
  for (int m = 1; m < 16; m <<= 1) v += __shfl_xor(v, m, 32);
  return v;
}

__global__ void cvt_f32_bf16(const float* __restrict__ in, bf16_t* __restrict__ out,
                             long n) {
  long i = (long)blockIdx.x * blockDim.x + threadIdx.x;
  long stride = (long)gridDim.x * blockDim.x;
  for (; i < n; i += stride) out[i] = (bf16_t)in[i];
}

// out[b][n][k] = (bf16) in[b][k][n]   (weights -> B-transposed [N][K] layout)
__global__ void cvt_transpose(const float* __restrict__ in, bf16_t* __restrict__ out,
                              int Kd, int Nd, long total) {
  long i = (long)blockIdx.x * blockDim.x + threadIdx.x;
  long stride = (long)gridDim.x * blockDim.x;
  const long KN = (long)Kd * Nd;
  for (; i < total; i += stride) {
    long b = i / KN;
    long rem = i - b * KN;
    int n = (int)(rem / Kd);
    int k = (int)(rem - (long)n * Kd);
    out[i] = (bf16_t)in[b * KN + (long)k * Nd + n];
  }
}

// C[M x N] = A[M x K](bf16,row-major,lda) * B[N x K](bf16,n-major,ldb)^T + bias
// grid: (N/64, M/64, batch); 128 threads (4 waves), each wave a 32x32 quadrant.
// Tiles are DMA'd into double-buffered LDS with global_load_async_to_lds_b128.
template <bool OUT_F32>
__global__ __launch_bounds__(128) void gemm_wmma(
    const bf16_t* __restrict__ A, long aBatch, int lda,
    const bf16_t* __restrict__ B, long bBatch, int ldb,
    void* __restrict__ Cout, long cBatch, int ldc,
    const float* __restrict__ bias, int K) {
  __shared__ bf16_t As[2][64][40];  // [buf][m][k], row stride 80B
  __shared__ bf16_t Bs[2][64][40];  // [buf][n][k]

  const int tid = threadIdx.x;
  const int w = tid >> 5, lane = tid & 31;
  const int hl = lane >> 4, l16 = lane & 15;
  const int wm = (w >> 1) * 32, wn = (w & 1) * 32;
  const int row0 = blockIdx.y * 64, col0 = blockIdx.x * 64;

  const bf16_t* Ab = A + (long)blockIdx.z * aBatch;
  const bf16_t* Bb = B + (long)blockIdx.z * bBatch;

  const unsigned asB = lds_off_of(&As[0][0][0]);
  const unsigned bsB = lds_off_of(&Bs[0][0][0]);
  constexpr unsigned BUFB = 64u * 40u * 2u;  // 5120 bytes per buffer

  auto issue_tile = [&](int k0, int buf) {
#pragma unroll
    for (int i = 0; i < 2; ++i) {
      int idx = tid * 2 + i;
      int r = idx >> 2, c8 = (idx & 3) * 8;
      unsigned loff = (unsigned)buf * BUFB + (unsigned)(r * 80 + c8 * 2);
      async_load_b128(asB + loff, Ab + (long)(row0 + r) * lda + k0 + c8);
      async_load_b128(bsB + loff, Bb + (long)(col0 + r) * ldb + k0 + c8);
    }
  };

  v8f acc[2][2];
#pragma unroll
  for (int i = 0; i < 2; ++i)
#pragma unroll
    for (int j = 0; j < 2; ++j) acc[i][j] = (v8f){};

  issue_tile(0, 0);
  wait_async0();
  __syncthreads();

  int buf = 0;
  for (int k0 = 0; k0 < K; k0 += 32) {
    if (k0 + 32 < K) issue_tile(k0 + 32, buf ^ 1);  // prefetch next tile (DMA)

    // Fragments per ISA VGPR layout (A and B: k = hl*8+j / 16+hl*8+j)
    v16bf af[2], bfr[2];
#pragma unroll
    for (int i = 0; i < 2; ++i) {
      V16U u;
      const bf16_t* p = &As[buf][wm + i * 16 + l16][0];
      u.h[0] = *(const v8bf*)(p + hl * 8);
      u.h[1] = *(const v8bf*)(p + 16 + hl * 8);
      af[i] = u.v;
    }
#pragma unroll
    for (int j = 0; j < 2; ++j) {
      V16U u;
      const bf16_t* p = &Bs[buf][wn + j * 16 + l16][0];
      u.h[0] = *(const v8bf*)(p + hl * 8);
      u.h[1] = *(const v8bf*)(p + 16 + hl * 8);
      bfr[j] = u.v;
    }
#pragma unroll
    for (int i = 0; i < 2; ++i)
#pragma unroll
      for (int j = 0; j < 2; ++j) acc[i][j] = wmma_bf16(af[i], bfr[j], acc[i][j]);

    wait_async0();
    __syncthreads();
    buf ^= 1;
  }

  // C layout: m = hl*8 + r, n = l16
#pragma unroll
  for (int i = 0; i < 2; ++i)
#pragma unroll
    for (int j = 0; j < 2; ++j) {
      const int col = col0 + wn + j * 16 + l16;
      const float badd = bias ? bias[col] : 0.f;
#pragma unroll
      for (int r = 0; r < 8; ++r) {
        const long row = row0 + wm + i * 16 + hl * 8 + r;
        const float vv = acc[i][j][r] + badd;
        if constexpr (OUT_F32) {
          float* C = (float*)Cout + (long)blockIdx.z * cBatch;
          C[row * ldc + col] = vv;
        } else {
          bf16_t* C = (bf16_t*)Cout + (long)blockIdx.z * cBatch;
          C[row * ldc + col] = (bf16_t)vv;
        }
      }
    }
}

// Causal flash attention. Q/K/V/Y are bf16 [T][D] with head at column h*64.
// grid: (T/64, H); block 128 (4 waves), wave w owns query rows qbase+w*16..+15.
__global__ __launch_bounds__(128) void attn_wmma(const bf16_t* __restrict__ Q,
                                                 const bf16_t* __restrict__ Kb,
                                                 const bf16_t* __restrict__ Vb,
                                                 bf16_t* __restrict__ Y) {
  __shared__ bf16_t kt[64][72];      // K tile row-major [key][feat] (async DMA)
  __shared__ bf16_t vt[64][72];      // V tile transposed [feat][key]
  __shared__ bf16_t pw[4][16][72];   // per-wave P staging (C-layout -> A-layout)

  const int tid = threadIdx.x;
  const int w = tid >> 5, lane = tid & 31;
  const int hl = lane >> 4, l16 = lane & 15;
  const int hc = blockIdx.y * HDn;
  const int qbase = blockIdx.x * 64;
  const float scale = 0.125f;  // 1/sqrt(64)
  const float NEG = -3.0e38f;

  const unsigned ktB = lds_off_of(&kt[0][0]);

  // Q fragments for this wave's 16 rows (kept in registers for all key tiles)
  v16bf aq[2];
#pragma unroll
  for (int c = 0; c < 2; ++c) {
    V16U u;
    const bf16_t* qp = Q + (long)(qbase + w * 16 + l16) * Dn + hc + c * 32;
    u.h[0] = *(const v8bf*)(qp + hl * 8);
    u.h[1] = *(const v8bf*)(qp + 16 + hl * 8);
    aq[c] = u.v;
  }

  v8f o[4];
#pragma unroll
  for (int nt = 0; nt < 4; ++nt) o[nt] = (v8f){};
  float mrow[8], lrow[8];
#pragma unroll
  for (int r = 0; r < 8; ++r) { mrow[r] = NEG; lrow[r] = 0.f; }

  const int ntiles = blockIdx.x + 1;  // causal: key tiles s0 <= qbase
  for (int t = 0; t < ntiles; ++t) {
    const int s0 = t * 64;
    __syncthreads();
    // K tile via async global->LDS DMA; V tile via vector load + transpose scatter
#pragma unroll
    for (int i = 0; i < 4; ++i) {
      int idx = tid * 4 + i;
      int r = idx >> 3, c8 = (idx & 7) * 8;
      async_load_b128(ktB + (unsigned)(r * 144 + c8 * 2),
                      Kb + (long)(s0 + r) * Dn + hc + c8);
      v8bf vv = *(const v8bf*)(Vb + (long)(s0 + r) * Dn + hc + c8);
#pragma unroll
      for (int jj = 0; jj < 8; ++jj) vt[c8 + jj][r] = vv[jj];
    }
    wait_async0();
    __syncthreads();

    // S = Q * K^T : 4 n-tiles x 2 k-chunks = 8 WMMA
    v8f s[4];
#pragma unroll
    for (int nt = 0; nt < 4; ++nt) {
      v8f a = (v8f){};
#pragma unroll
      for (int c = 0; c < 2; ++c) {
        V16U u;
        const bf16_t* p = &kt[nt * 16 + l16][c * 32 + hl * 8];
        u.h[0] = *(const v8bf*)p;
        u.h[1] = *(const v8bf*)(p + 16);
        a = wmma_bf16(aq[c], u.v, a);
      }
      s[nt] = a;
    }

    // online softmax (C-frag rows: m = hl*8 + r, key col = s0 + nt*16 + l16)
    float alpha[8];
#pragma unroll
    for (int r = 0; r < 8; ++r) {
      const int qi = qbase + w * 16 + hl * 8 + r;
      float mx = NEG;
#pragma unroll
      for (int nt = 0; nt < 4; ++nt) {
        const int kc = s0 + nt * 16 + l16;
        float sv = s[nt][r] * scale;
        sv = (kc > qi) ? NEG : sv;  // causal mask, branchless (EXEC stays full)
        s[nt][r] = sv;
        mx = fmaxf(mx, sv);
      }
      mx = rmax16(mx);
      const float nm = fmaxf(mrow[r], mx);
      alpha[r] = __expf(mrow[r] - nm);
      mrow[r] = nm;
      float ps = 0.f;
#pragma unroll
      for (int nt = 0; nt < 4; ++nt) {
        const float p = __expf(s[nt][r] - nm);
        s[nt][r] = p;
        ps += p;
      }
      ps = rsum16(ps);
      lrow[r] = lrow[r] * alpha[r] + ps;
    }

    // rescale O, stage P to LDS in row-major for A-fragment reload
#pragma unroll
    for (int nt = 0; nt < 4; ++nt)
#pragma unroll
      for (int r = 0; r < 8; ++r) {
        o[nt][r] = o[nt][r] * alpha[r];
        pw[w][hl * 8 + r][nt * 16 + l16] = (bf16_t)s[nt][r];
      }
    __syncthreads();

    // O += P * V : 4 dim-tiles x 2 key-chunks = 8 WMMA
    v16bf ap[2];
#pragma unroll
    for (int c = 0; c < 2; ++c) {
      V16U u;
      const bf16_t* p = &pw[w][l16][c * 32];
      u.h[0] = *(const v8bf*)(p + hl * 8);
      u.h[1] = *(const v8bf*)(p + 16 + hl * 8);
      ap[c] = u.v;
    }
#pragma unroll
    for (int nt = 0; nt < 4; ++nt)
#pragma unroll
      for (int c = 0; c < 2; ++c) {
        V16U u;
        const bf16_t* p = &vt[nt * 16 + l16][c * 32 + hl * 8];
        u.h[0] = *(const v8bf*)p;
        u.h[1] = *(const v8bf*)(p + 16);
        o[nt] = wmma_bf16(ap[c], u.v, o[nt]);
      }
  }

  // normalize rows and write y[t][h*64+d] (== transpose(0,2,1,3).reshape)
#pragma unroll
  for (int r = 0; r < 8; ++r) {
    const float inv = 1.f / lrow[r];
    const long row = qbase + w * 16 + hl * 8 + r;
#pragma unroll
    for (int nt = 0; nt < 4; ++nt)
      Y[row * Dn + hc + nt * 16 + l16] = (bf16_t)(o[nt][r] * inv);
  }
}

}  // namespace

extern "C" void kernel_launch(void* const* d_in, const int* in_sizes, int n_in,
                              void* d_out, int out_size, void* d_ws, size_t ws_size,
                              hipStream_t stream) {
  (void)in_sizes; (void)n_in; (void)out_size; (void)ws_size;
  const float* x   = (const float*)d_in[0];
  const float* q_a = (const float*)d_in[1];
  const float* q_b = (const float*)d_in[2];
  const float* k_a = (const float*)d_in[3];
  const float* k_b = (const float*)d_in[4];
  const float* v_w = (const float*)d_in[5];
  const float* v_b = (const float*)d_in[6];
  const float* o_w = (const float*)d_in[7];
  const float* o_b = (const float*)d_in[8];

  const long TD = (long)Tn * Dn;
  bf16_t* xb   = (bf16_t*)d_ws;   // bf16 hidden states [T][D]
  bf16_t* qmid = xb + TD;         // [T][H*R]   (head h at col h*64)
  bf16_t* kmid = qmid + TD;
  bf16_t* qh   = kmid + TD;       // [T][D] = [T][H*HD]
  bf16_t* kh   = qh + TD;
  bf16_t* vh   = kh + TD;
  bf16_t* yb   = vh + TD;         // attention output, bf16
  bf16_t* qaT  = yb + TD;                       // [H][R][D]
  bf16_t* kaT  = qaT + (long)Hn * Rn * Dn;      // [H][R][D]
  bf16_t* qbT  = kaT + (long)Hn * Rn * Dn;      // [H][HD][R]
  bf16_t* kbT  = qbT + (long)Hn * HDn * Rn;     // [H][HD][R]
  bf16_t* vwB  = kbT + (long)Hn * HDn * Rn;     // [D][D] (already n-major)
  bf16_t* owB  = vwB + (long)Dn * Dn;           // [D][D]

  // one-time bf16 conversions (weights into B[N][K] layout)
  cvt_f32_bf16<<<2048, 256, 0, stream>>>(x, xb, TD);
  cvt_transpose<<<512, 256, 0, stream>>>(q_a, qaT, Dn, Rn, (long)Hn * Dn * Rn);
  cvt_transpose<<<512, 256, 0, stream>>>(k_a, kaT, Dn, Rn, (long)Hn * Dn * Rn);
  cvt_transpose<<<64, 256, 0, stream>>>(q_b, qbT, Rn, HDn, (long)Hn * Rn * HDn);
  cvt_transpose<<<64, 256, 0, stream>>>(k_b, kbT, Rn, HDn, (long)Hn * Rn * HDn);
  cvt_f32_bf16<<<512, 256, 0, stream>>>(v_w, vwB, (long)Dn * Dn);
  cvt_f32_bf16<<<512, 256, 0, stream>>>(o_w, owB, (long)Dn * Dn);

  dim3 blk(128);
  // qmid[h] = xb * q_a[h]^T-layout   (M=T, N=R, K=D, batched over heads)
  gemm_wmma<false><<<dim3(Rn / 64, Tn / 64, Hn), blk, 0, stream>>>(
      xb, 0L, Dn, qaT, (long)Rn * Dn, Dn, qmid, 64L, Dn, nullptr, Dn);
  gemm_wmma<false><<<dim3(Rn / 64, Tn / 64, Hn), blk, 0, stream>>>(
      xb, 0L, Dn, kaT, (long)Rn * Dn, Dn, kmid, 64L, Dn, nullptr, Dn);
  // q[h] = qmid[h] * q_b[h]   (M=T, N=HD, K=R)
  gemm_wmma<false><<<dim3(HDn / 64, Tn / 64, Hn), blk, 0, stream>>>(
      qmid, 64L, Dn, qbT, (long)HDn * Rn, Rn, qh, 64L, Dn, nullptr, Rn);
  gemm_wmma<false><<<dim3(HDn / 64, Tn / 64, Hn), blk, 0, stream>>>(
      kmid, 64L, Dn, kbT, (long)HDn * Rn, Rn, kh, 64L, Dn, nullptr, Rn);
  // v = xb * v_w^T + v_b
  gemm_wmma<false><<<dim3(Dn / 64, Tn / 64, 1), blk, 0, stream>>>(
      xb, 0L, Dn, vwB, 0L, Dn, vh, 0L, Dn, v_b, Dn);
  // flash attention per head
  attn_wmma<<<dim3(Tn / 64, Hn), blk, 0, stream>>>(qh, kh, vh, yb);
  // out = y * o_w^T + o_b  (f32 output straight to d_out)
  gemm_wmma<true><<<dim3(Dn / 64, Tn / 64, 1), blk, 0, stream>>>(
      yb, 0L, Dn, owB, 0L, Dn, d_out, 0L, Dn, o_b, Dn);
}
